// CrossAttention_58351425683820
// MI455X (gfx1250) — matrix operand order repaired and checked by reading, not compiled, not run on hardware
//
#include <hip/hip_runtime.h>

#define BD 4
#define SD 2048
#define DD 1024
#define HH 16
#define HDD 64

typedef __attribute__((ext_vector_type(16))) __bf16 v16bf;
typedef __attribute__((ext_vector_type(8)))  float  v8f;
typedef int v4i __attribute__((vector_size(16)));

union Frag {
  v16bf v;
  uint4 q[2];
  unsigned short u[16];
};

__device__ __forceinline__ unsigned short f2bf(float f) {
  unsigned int u = __float_as_uint(f);
  u += 0x7fffu + ((u >> 16) & 1u);   // round to nearest even
  return (unsigned short)(u >> 16);
}

__device__ __forceinline__ v8f wmma_bf16(const Frag& a, const Frag& b, v8f c) {
  return __builtin_amdgcn_wmma_f32_16x16x32_bf16(false, a.v, false, b.v,
                                                 (short)0, c, false, false);
}

// ---- async global->LDS (CDNA5 GLOBAL_LOAD_ASYNC_TO_LDS_B128, ASYNCcnt) ----
#if __has_builtin(__builtin_amdgcn_global_load_async_to_lds_b128)
#define HAVE_ASYNC_LDS 1
#else
#define HAVE_ASYNC_LDS 0
#endif

__device__ __forceinline__ void cp_b128(const unsigned short* g, unsigned short* l) {
#if HAVE_ASYNC_LDS
  __builtin_amdgcn_global_load_async_to_lds_b128(
      (v4i*)(const_cast<unsigned short*>(g)),
      (__attribute__((address_space(3))) v4i*)(l), 0, 0);
#else
  *reinterpret_cast<uint4*>(l) = *reinterpret_cast<const uint4*>(g);
#endif
}
__device__ __forceinline__ void async_wait() {
#if HAVE_ASYNC_LDS
  asm volatile("s_wait_asynccnt 0" ::: "memory");
#endif
}

// ---- fragment loads ----
// A-fragment 16x32 (MxK): lane L: m=L&15, khalf=L>>4; K = khalf*8+{0..7}, 16+khalf*8+{0..7}
__device__ __forceinline__ void load_a_g(Frag& f, const unsigned short* __restrict__ A,
                                         int lda, int m0, int k0, int lane) {
  const unsigned short* p =
      A + (size_t)(m0 + (lane & 15)) * lda + k0 + ((lane >> 4) << 3);
  f.q[0] = *reinterpret_cast<const uint4*>(p);
  f.q[1] = *reinterpret_cast<const uint4*>(p + 16);
}
// B-fragment 32x16 (KxN) from NxK row-major: lane L: n=L&15, K = (L>>4)*8+{0..7}, 16+(L>>4)*8+{0..7}
__device__ __forceinline__ void load_b_lds(Frag& f, const unsigned short* sB,
                                           int stride, int n, int lane) {
  const unsigned short* p = sB + (n + (lane & 15)) * stride + ((lane >> 4) << 3);
  f.q[0] = *reinterpret_cast<const uint4*>(p);
  f.q[1] = *reinterpret_cast<const uint4*>(p + 16);
}
__device__ __forceinline__ void load_a_lds(Frag& f, const unsigned short* sA,
                                           int stride, int m, int lane) {
  const unsigned short* p = sA + (m + (lane & 15)) * stride + ((lane >> 4) << 3);
  f.q[0] = *reinterpret_cast<const uint4*>(p);
  f.q[1] = *reinterpret_cast<const uint4*>(p + 16);
}

__device__ __forceinline__ float half16_max(float v) {
  v = fmaxf(v, __shfl_xor(v, 1, 32));
  v = fmaxf(v, __shfl_xor(v, 2, 32));
  v = fmaxf(v, __shfl_xor(v, 4, 32));
  v = fmaxf(v, __shfl_xor(v, 8, 32));
  return v;
}
__device__ __forceinline__ float half16_sum(float v) {
  v += __shfl_xor(v, 1, 32);
  v += __shfl_xor(v, 2, 32);
  v += __shfl_xor(v, 4, 32);
  v += __shfl_xor(v, 8, 32);
  return v;
}

// ---------------- fp32 -> bf16 convert ----------------
__global__ void cvt_bf16_kernel(const float* __restrict__ in,
                                unsigned short* __restrict__ out, int n4) {
  int i = blockIdx.x * blockDim.x + threadIdx.x;
  if (i >= n4) return;
  float4 f = reinterpret_cast<const float4*>(in)[i];
  ushort4 o;
  o.x = f2bf(f.x); o.y = f2bf(f.y); o.z = f2bf(f.z); o.w = f2bf(f.w);
  reinterpret_cast<ushort4*>(out)[i] = o;
}

// stage 128x32 A tile + 128x32 B tile into padded LDS (row stride 40 elems)
__device__ __forceinline__ void stage_ab(const unsigned short* __restrict__ A,
                                         const unsigned short* __restrict__ W,
                                         unsigned short* sA, unsigned short* sB,
                                         int m0, int n0, int k0, int t) {
#pragma unroll
  for (int i = 0; i < 2; ++i) {
    int c = t + (i << 8);                    // 0..511 chunks of 8 elems
    int row = c >> 2, col = (c & 3) << 3;
    cp_b128(A + (size_t)(m0 + row) * DD + k0 + col, sA + row * 40 + col);
    cp_b128(W + (size_t)(n0 + row) * DD + k0 + col, sB + row * 40 + col);
  }
}

// inner product step: one B-fragment live at a time to cut VGPR pressure
__device__ __forceinline__ void gemm_step(v8f acc[2][4], const unsigned short* cA,
                                          const unsigned short* cB, int wm, int wn,
                                          int lane) {
  Frag a[2];
#pragma unroll
  for (int i = 0; i < 2; ++i) load_a_lds(a[i], cA, 40, wm * 32 + i * 16, lane);
#pragma unroll
  for (int j = 0; j < 4; ++j) {
    Frag bf;
    load_b_lds(bf, cB, 40, wn * 64 + j * 16, lane);
#pragma unroll
    for (int i = 0; i < 2; ++i) acc[i][j] = wmma_bf16(a[i], bf, acc[i][j]);
  }
}

// ---------------- fused QKV projection (LDS-staged NT GEMM) ----------------
__global__ __launch_bounds__(256) void qkv_gemm(
    const unsigned short* __restrict__ xb, const unsigned short* __restrict__ yb,
    const unsigned short* __restrict__ wq, const unsigned short* __restrict__ wk,
    const unsigned short* __restrict__ wv,
    const float* __restrict__ bq, const float* __restrict__ bk,
    const float* __restrict__ bv,
    unsigned short* __restrict__ qb, unsigned short* __restrict__ kb,
    unsigned short* __restrict__ vtb) {
  __shared__ alignas(16) unsigned short sA[2][128 * 40];
  __shared__ alignas(16) unsigned short sB[2][128 * 40];
  const int t = threadIdx.x;
  const int lane = t & 31, wave = t >> 5;
  const int wm = wave >> 1, wn = wave & 1;
  const int m0 = blockIdx.y * 128;
  const int n0 = blockIdx.x * 128;
  const int z = blockIdx.z;
  const unsigned short* A = (z == 1) ? yb : xb;
  const unsigned short* W = (z == 0) ? wq : (z == 1) ? wk : wv;
  const float* bias = (z == 0) ? bq : (z == 1) ? bk : bv;

  stage_ab(A, W, sA[0], sB[0], m0, n0, 0, t);

  v8f acc[2][4] = {};
  for (int ks = 0; ks < 32; ++ks) {
    const int buf = ks & 1;
    async_wait();
    __syncthreads();
    if (ks + 1 < 32)
      stage_ab(A, W, sA[buf ^ 1], sB[buf ^ 1], m0, n0, (ks + 1) * 32, t);
    gemm_step(acc, sA[buf], sB[buf], wm, wn, lane);
  }

  const int rbase = (lane >> 4) << 3, cc = lane & 15;
#pragma unroll
  for (int i = 0; i < 2; ++i)
#pragma unroll
    for (int j = 0; j < 4; ++j) {
      int gn = n0 + wn * 64 + j * 16 + cc;
      float bb = bias[gn];
#pragma unroll
      for (int r = 0; r < 8; ++r) {
        int gm = m0 + wm * 32 + i * 16 + rbase + r;
        unsigned short val = f2bf(acc[i][j][r] + bb);
        if (z == 0) {
          qb[(size_t)gm * DD + gn] = val;
        } else if (z == 1) {
          kb[(size_t)gm * DD + gn] = val;
        } else {
          int b_ = gm >> 11, s_ = gm & 2047, h_ = gn >> 6, d_ = gn & 63;
          vtb[((size_t)((b_ * HH + h_) * HDD + d_)) * SD + s_] = val;
        }
      }
    }
}

// stage 32x64 K tile (stride 72) + 64x32 Vt tile (stride 40) for one 32-key step
__device__ __forceinline__ void stage_kv(const unsigned short* __restrict__ Kp,
                                         const unsigned short* __restrict__ Vp,
                                         unsigned short* sK, unsigned short* sV,
                                         int key0, int t) {
  {
    int row = t >> 3, col = (t & 7) << 3;   // 32 rows x 64 elems
    cp_b128(Kp + (size_t)(key0 + row) * DD + col, sK + row * 72 + col);
  }
  {
    int row = t >> 2, col = (t & 3) << 3;   // 64 rows x 32 elems
    cp_b128(Vp + (size_t)row * SD + key0 + col, sV + row * 40 + col);
  }
}

// ---------------- flash attention: per-wave 16-query tile, shared K/V ----------------
__global__ __launch_bounds__(256) void attn_kernel(
    const unsigned short* __restrict__ qb, const unsigned short* __restrict__ kb,
    const unsigned short* __restrict__ vt, unsigned short* __restrict__ ctxb) {
  __shared__ alignas(16) unsigned short sK[2][32 * 72];
  __shared__ alignas(16) unsigned short sV[2][64 * 40];
  __shared__ alignas(16) unsigned short pstage[8][16 * 40];
  const int t = threadIdx.x;
  const int lane = t & 31;
  const int wave = t >> 5;
  const int qt = blockIdx.x * 8 + wave;   // 0..127 query tiles
  const int bh = blockIdx.y;              // 0..63
  const int b = bh >> 4, h = bh & 15;

  const unsigned short* Qp = qb + (size_t)b * SD * DD + h * HDD;
  const unsigned short* Kp = kb + (size_t)b * SD * DD + h * HDD;
  const unsigned short* Vp = vt + (size_t)bh * HDD * SD;

  Frag qf[2];
#pragma unroll
  for (int j = 0; j < 2; ++j) load_a_g(qf[j], Qp, DD, qt * 16, j * 32, lane);

  v8f O[4] = {};
  float mrow[8], lrow[8];
#pragma unroll
  for (int r = 0; r < 8; ++r) { mrow[r] = -3.0e38f; lrow[r] = 0.f; }

  unsigned short* st = pstage[wave];

  stage_kv(Kp, Vp, sK[0], sV[0], 0, t);

  for (int key0 = 0; key0 < SD; key0 += 32) {
    const int buf = (key0 >> 5) & 1;
    async_wait();
    __syncthreads();
    if (key0 + 32 < SD)
      stage_kv(Kp, Vp, sK[buf ^ 1], sV[buf ^ 1], key0 + 32, t);

    // S = Q K^T : two 16x16 score tiles over 32 keys, K-depth 64
    v8f s0 = {}, s1 = {};
#pragma unroll
    for (int j = 0; j < 2; ++j) {
      Frag kf;
      {
        const unsigned short* p =
            sK[buf] + (lane & 15) * 72 + j * 32 + ((lane >> 4) << 3);
        kf.q[0] = *reinterpret_cast<const uint4*>(p);
        kf.q[1] = *reinterpret_cast<const uint4*>(p + 16);
      }
      s0 = wmma_bf16(qf[j], kf, s0);
      {
        const unsigned short* p =
            sK[buf] + (16 + (lane & 15)) * 72 + j * 32 + ((lane >> 4) << 3);
        kf.q[0] = *reinterpret_cast<const uint4*>(p);
        kf.q[1] = *reinterpret_cast<const uint4*>(p + 16);
      }
      s1 = wmma_bf16(qf[j], kf, s1);
    }

    // online softmax (row stats over 16-lane halves)
    float alpha[8];
#pragma unroll
    for (int r = 0; r < 8; ++r) {
      float a = s0[r] * 0.125f;   // 1/sqrt(64)
      float c = s1[r] * 0.125f;
      float mx = half16_max(fmaxf(a, c));
      float mnew = fmaxf(mrow[r], mx);
      alpha[r] = __expf(mrow[r] - mnew);
      float p0 = __expf(a - mnew), p1 = __expf(c - mnew);
      s0[r] = p0; s1[r] = p1;
      lrow[r] = lrow[r] * alpha[r] + half16_sum(p0 + p1);
      mrow[r] = mnew;
    }
#pragma unroll
    for (int tt = 0; tt < 4; ++tt)
#pragma unroll
      for (int r = 0; r < 8; ++r) O[tt][r] *= alpha[r];

    // C-layout P -> LDS (16x32, padded stride 40) -> A-fragment layout
    {
      int rr = (lane >> 4) << 3, cc = lane & 15;
#pragma unroll
      for (int r = 0; r < 8; ++r) {
        st[(rr + r) * 40 + cc]      = f2bf(s0[r]);
        st[(rr + r) * 40 + 16 + cc] = f2bf(s1[r]);
      }
    }
    asm volatile("s_wait_dscnt 0" ::: "memory");
    Frag pf;
    {
      int m = lane & 15, kh = (lane >> 4) << 3;
      pf.q[0] = *reinterpret_cast<const uint4*>(&st[m * 40 + kh]);
      pf.q[1] = *reinterpret_cast<const uint4*>(&st[m * 40 + 16 + kh]);
    }
    asm volatile("s_wait_dscnt 0" ::: "memory");

    // ctx += P @ V
#pragma unroll
    for (int tt = 0; tt < 4; ++tt) {
      Frag vf;
      const unsigned short* p =
          sV[buf] + (tt * 16 + (lane & 15)) * 40 + ((lane >> 4) << 3);
      vf.q[0] = *reinterpret_cast<const uint4*>(p);
      vf.q[1] = *reinterpret_cast<const uint4*>(p + 16);
      O[tt] = wmma_bf16(pf, vf, O[tt]);
    }
  }

  // normalize + store ctx (bf16, [B*S, D] row major)
  float inv[8];
#pragma unroll
  for (int r = 0; r < 8; ++r) inv[r] = 1.0f / lrow[r];
  const int rbase = (lane >> 4) << 3, cc = lane & 15;
#pragma unroll
  for (int tt = 0; tt < 4; ++tt)
#pragma unroll
    for (int r = 0; r < 8; ++r) {
      int row = qt * 16 + rbase + r;
      int col = h * HDD + tt * 16 + cc;
      ctxb[((size_t)b * SD + row) * DD + col] = f2bf(O[tt][r] * inv[r]);
    }
}

// ---------------- output projection + bias + residual (LDS-staged) ----------------
__global__ __launch_bounds__(256) void out_gemm(
    const unsigned short* __restrict__ ctxb, const unsigned short* __restrict__ wd,
    const float* __restrict__ bd, const float* __restrict__ x,
    float* __restrict__ hbuf) {
  __shared__ alignas(16) unsigned short sA[2][128 * 40];
  __shared__ alignas(16) unsigned short sB[2][128 * 40];
  const int t = threadIdx.x;
  const int lane = t & 31, wave = t >> 5;
  const int wm = wave >> 1, wn = wave & 1;
  const int m0 = blockIdx.y * 128;
  const int n0 = blockIdx.x * 128;

  stage_ab(ctxb, wd, sA[0], sB[0], m0, n0, 0, t);

  v8f acc[2][4] = {};
  for (int ks = 0; ks < 32; ++ks) {
    const int buf = ks & 1;
    async_wait();
    __syncthreads();
    if (ks + 1 < 32)
      stage_ab(ctxb, wd, sA[buf ^ 1], sB[buf ^ 1], m0, n0, (ks + 1) * 32, t);
    gemm_step(acc, sA[buf], sB[buf], wm, wn, lane);
  }

  const int rbase = (lane >> 4) << 3, cc = lane & 15;
#pragma unroll
  for (int i = 0; i < 2; ++i)
#pragma unroll
    for (int j = 0; j < 4; ++j) {
      int gn = n0 + wn * 64 + j * 16 + cc;
      float bb = bd[gn];
#pragma unroll
      for (int r = 0; r < 8; ++r) {
        size_t idx = (size_t)(m0 + wm * 32 + i * 16 + rbase + r) * DD + gn;
        hbuf[idx] = acc[i][j][r] + bb + x[idx];
      }
    }
}

// ---------------- row LayerNorm ----------------
__global__ __launch_bounds__(256) void ln_kernel(const float* __restrict__ hbuf,
                                                 const float* __restrict__ gamma,
                                                 const float* __restrict__ beta,
                                                 float* __restrict__ out) {
  __shared__ float rs[256], rss[256];
  const int t = threadIdx.x;
  const size_t row = blockIdx.x;
  const float* hr = hbuf + row * DD;
  float vals[4];
  float s = 0.f, ss = 0.f;
#pragma unroll
  for (int i = 0; i < 4; ++i) {
    float f = hr[t + i * 256];
    vals[i] = f; s += f; ss += f * f;
  }
  rs[t] = s; rss[t] = ss;
  __syncthreads();
  for (int o = 128; o > 0; o >>= 1) {
    if (t < o) { rs[t] += rs[t + o]; rss[t] += rss[t + o]; }
    __syncthreads();
  }
  float mean = rs[0] * (1.0f / DD);
  float var  = rss[0] * (1.0f / DD) - mean * mean;
  float invs = rsqrtf(var + 1e-12f);
  float* orow = out + row * DD;
#pragma unroll
  for (int i = 0; i < 4; ++i) {
    int c = t + i * 256;
    orow[c] = gamma[c] * ((vals[i] - mean) * invs) + beta[c];
  }
}

extern "C" void kernel_launch(void* const* d_in, const int* in_sizes, int n_in,
                              void* d_out, int out_size, void* d_ws, size_t ws_size,
                              hipStream_t stream) {
  (void)in_sizes; (void)n_in; (void)out_size; (void)ws_size;
  const float* x     = (const float*)d_in[0];
  const float* y     = (const float*)d_in[1];
  const float* Wq    = (const float*)d_in[2];
  const float* bq    = (const float*)d_in[3];
  const float* Wk    = (const float*)d_in[4];
  const float* bk    = (const float*)d_in[5];
  const float* Wv    = (const float*)d_in[6];
  const float* bv    = (const float*)d_in[7];
  const float* Wd    = (const float*)d_in[8];
  const float* bd    = (const float*)d_in[9];
  const float* gamma = (const float*)d_in[10];
  const float* beta  = (const float*)d_in[11];
  float* out = (float*)d_out;

  char* base = (char*)d_ws;
  size_t off = 0;
  auto take = [&](size_t bytes) {
    char* p = base + off;
    off += (bytes + 255) & ~(size_t)255;
    return p;
  };
  const size_t NX = (size_t)BD * SD * DD;  // 8,388,608
  const size_t NW = (size_t)DD * DD;       // 1,048,576
  unsigned short* xb   = (unsigned short*)take(NX * 2);
  unsigned short* yb   = (unsigned short*)take(NX * 2);
  unsigned short* wqb  = (unsigned short*)take(NW * 2);
  unsigned short* wkb  = (unsigned short*)take(NW * 2);
  unsigned short* wvb  = (unsigned short*)take(NW * 2);
  unsigned short* wdb  = (unsigned short*)take(NW * 2);
  unsigned short* qb   = (unsigned short*)take(NX * 2);
  unsigned short* kbuf = (unsigned short*)take(NX * 2);
  unsigned short* vtb  = (unsigned short*)take(NX * 2);
  unsigned short* ctxb = (unsigned short*)take(NX * 2);
  float* hbuf = (float*)take(NX * 4);

  auto cvt = [&](const float* src, unsigned short* dst, size_t n) {
    int n4 = (int)(n / 4);
    cvt_bf16_kernel<<<dim3((n4 + 255) / 256), dim3(256), 0, stream>>>(src, dst, n4);
  };
  cvt(x, xb, NX);
  cvt(y, yb, NX);
  cvt(Wq, wqb, NW);
  cvt(Wk, wkb, NW);
  cvt(Wv, wvb, NW);
  cvt(Wd, wdb, NW);

  qkv_gemm<<<dim3(DD / 128, (BD * SD) / 128, 3), dim3(256), 0, stream>>>(
      xb, yb, wqb, wkb, wvb, bq, bk, bv, qb, kbuf, vtb);

  attn_kernel<<<dim3(SD / 128, BD * HH), dim3(256), 0, stream>>>(qb, kbuf, vtb, ctxb);

  out_gemm<<<dim3(DD / 128, (BD * SD) / 128), dim3(256), 0, stream>>>(
      ctxb, wdb, bd, x, hbuf);

  ln_kernel<<<dim3(BD * SD), dim3(256), 0, stream>>>(hbuf, gamma, beta, out);
}
